// GalerkinTransformer_20486994002454
// MI455X (gfx1250) — matrix operand order repaired and checked by reading, not compiled, not run on hardware
//
#include <hip/hip_runtime.h>
#include <hip/hip_bf16.h>

typedef __attribute__((ext_vector_type(16))) _Float16 v16h;
typedef __attribute__((ext_vector_type(8)))  _Float16 v8h;
typedef __attribute__((ext_vector_type(8)))  float    v8f;
typedef int v4i __attribute__((vector_size(16)));

#define TOKENS 32768
#define NSEQ   8192
#define HID    256
#define NHEAD  8
#define DHEAD  32

// ---------------- CDNA5 async global->LDS support ---------------------------
#if defined(__gfx1250__) && __has_builtin(__builtin_amdgcn_global_load_async_to_lds_b128)
#define ASYNC_COPY 1
#else
#define ASYNC_COPY 0
#endif

#if ASYNC_COPY
__device__ __forceinline__ void g2l_async_b128(const _Float16* g, _Float16* l)
{
    __builtin_amdgcn_global_load_async_to_lds_b128(
        (__attribute__((address_space(1))) v4i*)(g),
        (__attribute__((address_space(3))) v4i*)(l), 0, 0);
}
#if __has_builtin(__builtin_amdgcn_s_wait_asynccnt)
#define WAIT_ASYNC0() __builtin_amdgcn_s_wait_asynccnt(0)
#else
#define WAIT_ASYNC0() asm volatile("s_wait_asynccnt 0" ::: "memory")
#endif
#define WG_BARRIER() asm volatile("s_barrier_signal -1\ns_barrier_wait -1" ::: "memory")
#endif

// ---------------------------------------------------------------------------
// Preprocess: fx = pre2(gelu(pre1(x))) + placeholder      (per-token, f32)
// ---------------------------------------------------------------------------
__global__ __launch_bounds__(256)
void k_pre(const float* __restrict__ x,  const float* __restrict__ w1,
           const float* __restrict__ b1, const float* __restrict__ w2,
           const float* __restrict__ b2, const float* __restrict__ ph,
           float* __restrict__ fx)
{
    __shared__ float sx[6];
    __shared__ float hid[512];
    const int t   = blockIdx.x;
    const int tid = threadIdx.x;
    if (tid < 6) sx[tid] = x[(size_t)t * 6 + tid];
    __syncthreads();
#pragma unroll
    for (int it = 0; it < 2; ++it) {
        const int u = tid + it * 256;
        float s = b1[u];
#pragma unroll
        for (int i = 0; i < 6; ++i) s += w1[u * 6 + i] * sx[i];
        hid[u] = 0.5f * s * (1.0f + erff(s * 0.70710678118f));   // exact GELU
    }
    __syncthreads();
    float s = b2[tid];
    const float* wr = w2 + (size_t)tid * 512;
    for (int i = 0; i < 512; ++i) s += wr[i] * hid[i];
    fx[(size_t)t * HID + tid] = s + ph[tid];
}

// ---------------------------------------------------------------------------
// LayerNorm over 256 channels, one wave32 per token, f32 in -> f16 out
// ---------------------------------------------------------------------------
__global__ __launch_bounds__(256)
void k_ln(const float* __restrict__ x, const float* __restrict__ g,
          const float* __restrict__ b, _Float16* __restrict__ y)
{
    const int lane = threadIdx.x & 31;
    const int wave = threadIdx.x >> 5;
    const int t    = blockIdx.x * 8 + wave;
    const float* xr = x + (size_t)t * HID;
    float v[8];
    float s = 0.f;
#pragma unroll
    for (int j = 0; j < 8; ++j) { v[j] = xr[j * 32 + lane]; s += v[j]; }
#pragma unroll
    for (int m = 16; m >= 1; m >>= 1) s += __shfl_xor(s, m, 32);
    const float mu = s * (1.0f / 256.0f);
    float q = 0.f;
#pragma unroll
    for (int j = 0; j < 8; ++j) { const float d = v[j] - mu; q += d * d; }
#pragma unroll
    for (int m = 16; m >= 1; m >>= 1) q += __shfl_xor(q, m, 32);
    const float inv = rsqrtf(q * (1.0f / 256.0f) + 1e-5f);
#pragma unroll
    for (int j = 0; j < 8; ++j) {
        const int c = j * 32 + lane;
        y[(size_t)t * HID + c] = (_Float16)((v[j] - mu) * inv * g[c] + b[c]);
    }
}

__global__ __launch_bounds__(256)
void k_cvt(const float* __restrict__ in, _Float16* __restrict__ out, int n)
{
    const int i = blockIdx.x * 256 + threadIdx.x;
    if (i < n) out[i] = (_Float16)in[i];
}

__global__ __launch_bounds__(256)
void k_zero(float* __restrict__ p, int n)
{
    const int i = blockIdx.x * 256 + threadIdx.x;
    if (i < n) p[i] = 0.f;
}

// ---------------------------------------------------------------------------
// WMMA GEMM:  C[M,N] = A[M,K](f16) * W[N,K]^T(f16) + bias
//   EPI 0: store f32 | EPI 1: exact GELU -> f16 | EPI 2: += resid -> f32
// Block 256 thr (8 waves). Tile BM=128 BN=128 BK=32.
// Wave w owns C rows [w*16, +16) x all 128 cols -> 8 accumulators.
// Async global->LDS double-buffer pipeline: one barrier per K-step.
// ---------------------------------------------------------------------------
#define BM 128
#define BN 128
#define BK 32
#define LDA 40   // halves: 80B row stride -> conflict-free b128 fragment reads
#define LDB 40

template <int EPI>
__global__ __launch_bounds__(256)
void k_gemm(const _Float16* __restrict__ A, const _Float16* __restrict__ W,
            const float* __restrict__ bias, void* __restrict__ Cout,
            const float* __restrict__ resid, int M, int N, int K)
{
    __shared__ _Float16 As[2][BM * LDA];
    __shared__ _Float16 Bs[2][BN * LDB];
    const int tid  = threadIdx.x;
    const int lane = tid & 31;
    const int wave = tid >> 5;     // 0..7
    const int hl   = lane & 15;
    const int hi   = lane >> 4;
    const int rowBase = blockIdx.y * BM;
    const int colBase = blockIdx.x * BN;

    // this thread's two 16B chunks of each 128x32-half tile
    const int r0  = tid >> 2;                 // 0..63
    const int r1  = r0 + 64;                  // 64..127
    const int ck  = (tid & 3) << 3;           // 0,8,16,24 halves

    v8f acc[8] = {};
    const int ntiles = K / BK;

    const int arow = wave * 16 + hl;
    const int akb  = hi ? 8 : 0;
    const int bkb  = hi ? 8 : 0;

    auto compute = [&](const _Float16* as, const _Float16* bs) {
        const v8h alo = *(const v8h*)(as + arow * LDA + akb);
        const v8h ahi = *(const v8h*)(as + arow * LDA + akb + 16);
        v16h af;
#pragma unroll
        for (int i = 0; i < 8; ++i) { af[i] = alo[i]; af[i + 8] = ahi[i]; }
#pragma unroll
        for (int nt = 0; nt < 8; ++nt) {
            const int bcol = nt * 16 + hl;
            const v8h blo = *(const v8h*)(bs + bcol * LDB + bkb);
            const v8h bhi = *(const v8h*)(bs + bcol * LDB + bkb + 16);
            v16h bf;
#pragma unroll
            for (int i = 0; i < 8; ++i) { bf[i] = blo[i]; bf[i + 8] = bhi[i]; }
            acc[nt] = __builtin_amdgcn_wmma_f32_16x16x32_f16(
                false, af, false, bf, (short)0, acc[nt], false, false);
        }
    };

#if ASYNC_COPY
    auto issue = [&](int t, int buf) {
        const size_t kof = (size_t)t * BK;
        g2l_async_b128(A + (size_t)(rowBase + r0) * K + kof + ck, &As[buf][r0 * LDA + ck]);
        g2l_async_b128(A + (size_t)(rowBase + r1) * K + kof + ck, &As[buf][r1 * LDA + ck]);
        g2l_async_b128(W + (size_t)(colBase + r0) * K + kof + ck, &Bs[buf][r0 * LDB + ck]);
        g2l_async_b128(W + (size_t)(colBase + r1) * K + kof + ck, &Bs[buf][r1 * LDB + ck]);
    };
    issue(0, 0);
    for (int t = 0; t < ntiles; ++t) {
        const int cur = t & 1;
        WAIT_ASYNC0();              // our async writes for buf[cur] have landed
        WG_BARRIER();               // everyone's writes landed; prev compute done
        if (t + 1 < ntiles) issue(t + 1, cur ^ 1);   // overlaps compute below
        compute(As[cur], Bs[cur]);
    }
#else
    for (int t = 0; t < ntiles; ++t) {
        const size_t kof = (size_t)t * BK;
        const uint4 a0 = *(const uint4*)(A + (size_t)(rowBase + r0) * K + kof + ck);
        const uint4 a1 = *(const uint4*)(A + (size_t)(rowBase + r1) * K + kof + ck);
        const uint4 b0 = *(const uint4*)(W + (size_t)(colBase + r0) * K + kof + ck);
        const uint4 b1 = *(const uint4*)(W + (size_t)(colBase + r1) * K + kof + ck);
        __syncthreads();            // previous tile fully consumed
        *(uint4*)(&As[0][r0 * LDA + ck]) = a0;
        *(uint4*)(&As[0][r1 * LDA + ck]) = a1;
        *(uint4*)(&Bs[0][r0 * LDB + ck]) = b0;
        *(uint4*)(&Bs[0][r1 * LDB + ck]) = b1;
        __syncthreads();
        compute(As[0], Bs[0]);
    }
#endif

    // ---- epilogue: acc VGPR p = (M = p + hi*8, N = hl) of each 16x16 tile ----
#pragma unroll
    for (int nt = 0; nt < 8; ++nt) {
        const int col = colBase + nt * 16 + hl;
        const float bv = bias[col];
#pragma unroll
        for (int p = 0; p < 8; ++p) {
            const int row = rowBase + wave * 16 + p + hi * 8;
            const size_t idx = (size_t)row * N + col;
            float v = acc[nt][p] + bv;
            if (EPI == 0) {
                ((float*)Cout)[idx] = v;
            } else if (EPI == 1) {
                v = 0.5f * v * (1.0f + erff(v * 0.70710678118f));
                ((_Float16*)Cout)[idx] = (_Float16)v;
            } else {
                ((float*)Cout)[idx] = v + resid[idx];
            }
        }
    }
}

// ---------------------------------------------------------------------------
// Column softmax stats for k: per (b, channel) max and sum(exp(.-max)) over N
// ---------------------------------------------------------------------------
__global__ __launch_bounds__(256)
void k_colstats(const float* __restrict__ kraw, float* __restrict__ mx,
                float* __restrict__ sm)
{
    __shared__ float red[256];
    const int bc = blockIdx.x;          // b*256 + c
    const int b = bc >> 8, c = bc & 255;
    const int tid = threadIdx.x;
    const float* base = kraw + (size_t)b * NSEQ * HID + c;
    float m = -1e30f;
    for (int n = tid; n < NSEQ; n += 256) m = fmaxf(m, base[(size_t)n * HID]);
    red[tid] = m; __syncthreads();
    for (int s = 128; s >= 1; s >>= 1) {
        if (tid < s) red[tid] = fmaxf(red[tid], red[tid + s]);
        __syncthreads();
    }
    const float M = red[0]; __syncthreads();
    float sum = 0.f;
    for (int n = tid; n < NSEQ; n += 256) sum += expf(base[(size_t)n * HID] - M);
    red[tid] = sum; __syncthreads();
    for (int s = 128; s >= 1; s >>= 1) {
        if (tid < s) red[tid] += red[tid + s];
        __syncthreads();
    }
    if (tid == 0) { mx[bc] = M; sm[bc] = red[0]; }
}

// ---------------------------------------------------------------------------
// ctx[b,h] += softmax_col(k)^T v  over a 128-row chunk  (32x32 per (b,h))
// ---------------------------------------------------------------------------
__global__ __launch_bounds__(256)
void k_ctx(const float* __restrict__ kraw, const float* __restrict__ vraw,
           const float* __restrict__ mx, const float* __restrict__ sm,
           float* __restrict__ ctx)
{
    __shared__ float ke[8][32];
    __shared__ float vv[8][32];
    const int CH = NSEQ / 128;
    int id = blockIdx.x;
    const int ch = id % CH; id /= CH;
    const int h  = id % NHEAD; id /= NHEAD;
    const int b  = id;
    const int tid = threadIdx.x;
    const int e  = tid & 31;
    const int dg = tid >> 5;            // d = dg*4 + j
    float acc[4] = {0.f, 0.f, 0.f, 0.f};
    const size_t tokBase = (size_t)b * NSEQ + (size_t)ch * 128;

    for (int r0 = 0; r0 < 128; r0 += 8) {
        const int r = tid >> 5, c = tid & 31;
        const size_t off = (tokBase + r0 + r) * HID + h * DHEAD + c;
        const int sc = b * HID + h * DHEAD + c;
        ke[r][c] = expf(kraw[off] - mx[sc]) / sm[sc];
        vv[r][c] = vraw[off];
        __syncthreads();
#pragma unroll
        for (int rr = 0; rr < 8; ++rr) {
            const float ve = vv[rr][e];
#pragma unroll
            for (int j = 0; j < 4; ++j) acc[j] += ke[rr][dg * 4 + j] * ve;
        }
        __syncthreads();
    }
#pragma unroll
    for (int j = 0; j < 4; ++j) {
        const int d = dg * 4 + j;
        atomicAdd(&ctx[(((size_t)b * NHEAD + h) * DHEAD + d) * DHEAD + e], acc[j]);
    }
}

// ---------------------------------------------------------------------------
// Per-token: q feature-softmax (per head, wave-local) then (q @ ctx)/N -> f16
// ---------------------------------------------------------------------------
__global__ __launch_bounds__(256)
void k_qattn(const float* __restrict__ qraw, const float* __restrict__ ctx,
             _Float16* __restrict__ out)
{
    __shared__ float sq[HID];
    const int t   = blockIdx.x;
    const int b   = t >> 13;
    const int tid = threadIdx.x;
    sq[tid] = qraw[(size_t)t * HID + tid];
    __syncthreads();
    const int lane = tid & 31, w = tid >> 5;    // wave w == head w
    const float val = sq[w * 32 + lane];
    float m = val;
    for (int s = 16; s >= 1; s >>= 1) m = fmaxf(m, __shfl_xor(m, s, 32));
    const float e = expf(val - m);
    float sum = e;
    for (int s = 16; s >= 1; s >>= 1) sum += __shfl_xor(sum, s, 32);
    __syncthreads();
    sq[w * 32 + lane] = e / sum;
    __syncthreads();
    const int h = tid >> 5, eo = tid & 31;
    const float* cx = ctx + (((size_t)b * NHEAD + h) * DHEAD) * DHEAD + eo;
    float s = 0.f;
#pragma unroll
    for (int d = 0; d < DHEAD; ++d) s += sq[h * 32 + d] * cx[d * DHEAD];
    out[(size_t)t * HID + tid] = (_Float16)(s * (1.0f / (float)NSEQ));
}

// ---------------------------------------------------------------------------
// Final head: out[t,o] = dot256(ln3(fx)[t], head.w[o]) + head.b[o]
// ---------------------------------------------------------------------------
__global__ __launch_bounds__(256)
void k_head(const _Float16* __restrict__ xn, const float* __restrict__ w,
            const float* __restrict__ b, float* __restrict__ out)
{
    const int id = blockIdx.x * 256 + threadIdx.x;  // over TOKENS*4
    const int t = id >> 2, o = id & 3;
    float s = b[o];
    const _Float16* xr = xn + (size_t)t * HID;
    const float* wr = w + o * HID;
    for (int i = 0; i < HID; ++i) s += (float)xr[i] * wr[i];
    out[id] = s;
}

// ---------------------------------------------------------------------------
// Host launch
// ---------------------------------------------------------------------------
extern "C" void kernel_launch(void* const* d_in, const int* in_sizes, int n_in,
                              void* d_out, int out_size, void* d_ws, size_t ws_size,
                              hipStream_t stream)
{
    (void)in_sizes; (void)n_in; (void)out_size; (void)ws_size;
    const float* X = (const float*)d_in[0];

    char* wp = (char*)d_ws;
    auto take = [&](size_t bytes) -> void* {
        void* p = (void*)wp;
        wp += (bytes + 255) & ~(size_t)255;
        return p;
    };
    float*    fx     = (float*)   take((size_t)TOKENS * HID * 4);
    float*    qraw   = (float*)   take((size_t)TOKENS * HID * 4);
    float*    kraw   = (float*)   take((size_t)TOKENS * HID * 4);
    float*    vraw   = (float*)   take((size_t)TOKENS * HID * 4);
    _Float16* xq16   = (_Float16*)take((size_t)TOKENS * HID * 2);
    _Float16* xkv16  = (_Float16*)take((size_t)TOKENS * HID * 2);
    _Float16* attn16 = (_Float16*)take((size_t)TOKENS * HID * 2);
    _Float16* h16    = (_Float16*)take((size_t)TOKENS * 1024 * 2);
    _Float16* wf16   = (_Float16*)take((size_t)1024 * 256 * 2);
    float*    mxb    = (float*)   take(1024 * 4);
    float*    smb    = (float*)   take(1024 * 4);
    float*    ctx    = (float*)   take((size_t)4 * NHEAD * DHEAD * DHEAD * 4);

    k_pre<<<TOKENS, 256, 0, stream>>>(X,
        (const float*)d_in[1], (const float*)d_in[2],
        (const float*)d_in[3], (const float*)d_in[4],
        (const float*)d_in[5], fx);

    const dim3 g256(HID / BN, TOKENS / BM);     // N=256  -> (2,256)
    const dim3 g1024(1024 / BN, TOKENS / BM);   // N=1024 -> (8,256)

    for (int l = 0; l < 8; ++l) {
        const int base = 6 + l * 18;
        const float* ln1g  = (const float*)d_in[base + 0];
        const float* ln1b  = (const float*)d_in[base + 1];
        const float* ln1ag = (const float*)d_in[base + 2];
        const float* ln1ab = (const float*)d_in[base + 3];
        const float* qW = (const float*)d_in[base + 4];
        const float* qB = (const float*)d_in[base + 5];
        const float* kW = (const float*)d_in[base + 6];
        const float* kB = (const float*)d_in[base + 7];
        const float* vW = (const float*)d_in[base + 8];
        const float* vB = (const float*)d_in[base + 9];
        const float* oW = (const float*)d_in[base + 10];
        const float* oB = (const float*)d_in[base + 11];
        const float* ln2g = (const float*)d_in[base + 12];
        const float* ln2b = (const float*)d_in[base + 13];
        const float* m1W = (const float*)d_in[base + 14];
        const float* m1B = (const float*)d_in[base + 15];
        const float* m2W = (const float*)d_in[base + 16];
        const float* m2B = (const float*)d_in[base + 17];

        k_ln<<<TOKENS / 8, 256, 0, stream>>>(fx, ln1g,  ln1b,  xq16);
        k_ln<<<TOKENS / 8, 256, 0, stream>>>(fx, ln1ag, ln1ab, xkv16);

        k_cvt<<<256, 256, 0, stream>>>(qW, wf16, 65536);
        k_gemm<0><<<g256, 256, 0, stream>>>(xq16, wf16, qB, (void*)qraw, nullptr,
                                            TOKENS, 256, 256);
        k_cvt<<<256, 256, 0, stream>>>(kW, wf16, 65536);
        k_gemm<0><<<g256, 256, 0, stream>>>(xkv16, wf16, kB, (void*)kraw, nullptr,
                                            TOKENS, 256, 256);
        k_cvt<<<256, 256, 0, stream>>>(vW, wf16, 65536);
        k_gemm<0><<<g256, 256, 0, stream>>>(xkv16, wf16, vB, (void*)vraw, nullptr,
                                            TOKENS, 256, 256);

        k_colstats<<<1024, 256, 0, stream>>>(kraw, mxb, smb);
        k_zero<<<128, 256, 0, stream>>>(ctx, 4 * NHEAD * DHEAD * DHEAD);
        k_ctx<<<4 * NHEAD * (NSEQ / 128), 256, 0, stream>>>(kraw, vraw, mxb, smb, ctx);
        k_qattn<<<TOKENS, 256, 0, stream>>>(qraw, ctx, attn16);

        k_cvt<<<256, 256, 0, stream>>>(oW, wf16, 65536);
        k_gemm<2><<<g256, 256, 0, stream>>>(attn16, wf16, oB, (void*)fx, fx,
                                            TOKENS, 256, 256);

        k_ln<<<TOKENS / 8, 256, 0, stream>>>(fx, ln2g, ln2b, xkv16);
        k_cvt<<<1024, 256, 0, stream>>>(m1W, wf16, 262144);
        k_gemm<1><<<g1024, 256, 0, stream>>>(xkv16, wf16, m1B, (void*)h16, nullptr,
                                             TOKENS, 1024, 256);
        k_cvt<<<1024, 256, 0, stream>>>(m2W, wf16, 262144);
        k_gemm<2><<<g256, 256, 0, stream>>>(h16, wf16, m2B, (void*)fx, fx,
                                            TOKENS, 256, 1024);
    }

    k_ln<<<TOKENS / 8, 256, 0, stream>>>(fx, (const float*)d_in[150],
                                         (const float*)d_in[151], xkv16);
    k_head<<<(TOKENS * 4) / 256, 256, 0, stream>>>(xkv16, (const float*)d_in[152],
                                                   (const float*)d_in[153],
                                                   (float*)d_out);
}
